// DilatedRNN_40286793237072
// MI455X (gfx1250) — matrix operand order repaired and checked by reading, not compiled, not run on hardware
//
#include <hip/hip_runtime.h>
#include <hip/hip_bf16.h>

// ---------------------------------------------------------------------------
// Dilated 3-layer GRU (B=256, T=168, H=512) for MI455X / gfx1250.
//
// Latency-bound serial scan; all weights (split-bf16 hi/lo, ~16MB) are
// L2-resident (192MB L2). Per step:
//  - fused GEMM kernel per layer (gi [+ gh only when t%dilation==0, decided
//    on host]) using V_WMMA_F32_16X16X32_BF16, split-bf16 3-pass for ~fp32
//    accuracy. Each wave computes a 32x32 C tile: 16 b128 loads / 12 wmma.
//  - fused epilogue per layer: GRU gate math, carry update, hi/lo re-split,
//    output-head dot (block reduce), h2 variance for aux.
// ---------------------------------------------------------------------------

typedef __bf16 bf16;
typedef __attribute__((ext_vector_type(16))) __bf16 v16bf;
typedef __attribute__((ext_vector_type(8)))  __bf16 v8bf;
typedef __attribute__((ext_vector_type(8)))  float  v8f;

#define B_   256
#define T_   168
#define H_   512
#define G_   1536   // 3*H
#define NIN_ 64

// ---------------------------------------------------------------------------
// Fragment loaders per CDNA5 ISA §7.12.2 (wave32).
// A 16x32 bf16 (MxK): lane r=lane&15 holds row M=r; lane group g=lane>>4.
//   vector elems 0..7  = K = 8g + 0..7      (one 16B chunk)
//   vector elems 8..15 = K = 16 + 8g + 0..7 (one 16B chunk)
// B 32x16 bf16 (KxN): lane holds column N=r; same K order as A
//   (two 16B chunks along the weight's K-major row -> no transpose needed).
// C/D f32 16x16: VGPR v -> M = v + 8g, N = r.
// ---------------------------------------------------------------------------
__device__ __forceinline__ v16bf load_afrag(const bf16* __restrict__ A,
                                            long long lda, int m0, int k0, int lane) {
  int g = lane >> 4, r = lane & 15;
  const bf16* p = A + (long long)(m0 + r) * lda + k0;
  v8bf c0 = *(const v8bf*)(p + 8 * g);
  v8bf c1 = *(const v8bf*)(p + 16 + 8 * g);
  v16bf a;
#pragma unroll
  for (int i = 0; i < 8; ++i) { a[i] = c0[i]; a[8 + i] = c1[i]; }
  return a;
}

__device__ __forceinline__ v16bf load_bfrag(const bf16* __restrict__ W,
                                            long long K, int n0, int k0, int lane) {
  int g = lane >> 4, r = lane & 15;
  const bf16* p = W + (long long)(n0 + r) * K + k0 + 8 * g;
  v8bf c0 = *(const v8bf*)(p);
  v8bf c1 = *(const v8bf*)(p + 16);
  v16bf b;
#pragma unroll
  for (int i = 0; i < 8; ++i) { b[i] = c0[i]; b[8 + i] = c1[i]; }
  return b;
}

__device__ __forceinline__ v8f wmma_bf16(v16bf a, v16bf b, v8f c) {
  // (neg_a, A, neg_b, B, c_mod, C, reuse_a, reuse_b)
  return __builtin_amdgcn_wmma_f32_16x16x32_bf16(false, a, false, b, (short)0, c,
                                                 false, false);
}

// One wave computes a 32x32 output tile (2 M-tiles x 2 N-tiles) with
// split-bf16: acc += Al*Bh + Ah*Bl + Ah*Bh (~fp32 accuracy, f32 accumulate).
// 16 b128 loads / 12 wmma per K-step.
__device__ __forceinline__ void gemm_tile22(const bf16* __restrict__ Ah,
                                            const bf16* __restrict__ Al,
                                            long long lda,
                                            const bf16* __restrict__ Wh,
                                            const bf16* __restrict__ Wl,
                                            int K, float* __restrict__ O,
                                            int m0, int n0, int lane) {
  v8f a00 = {0.f, 0.f, 0.f, 0.f, 0.f, 0.f, 0.f, 0.f};
  v8f a01 = a00, a10 = a00, a11 = a00;
#pragma unroll 2
  for (int k0 = 0; k0 < K; k0 += 32) {
    v16bf ah0 = load_afrag(Ah, lda, m0,      k0, lane);
    v16bf al0 = load_afrag(Al, lda, m0,      k0, lane);
    v16bf ah1 = load_afrag(Ah, lda, m0 + 16, k0, lane);
    v16bf al1 = load_afrag(Al, lda, m0 + 16, k0, lane);
    v16bf bh0 = load_bfrag(Wh, K, n0,      k0, lane);
    v16bf bl0 = load_bfrag(Wl, K, n0,      k0, lane);
    v16bf bh1 = load_bfrag(Wh, K, n0 + 16, k0, lane);
    v16bf bl1 = load_bfrag(Wl, K, n0 + 16, k0, lane);
    a00 = wmma_bf16(al0, bh0, a00);
    a00 = wmma_bf16(ah0, bl0, a00);
    a00 = wmma_bf16(ah0, bh0, a00);
    a01 = wmma_bf16(al0, bh1, a01);
    a01 = wmma_bf16(ah0, bl1, a01);
    a01 = wmma_bf16(ah0, bh1, a01);
    a10 = wmma_bf16(al1, bh0, a10);
    a10 = wmma_bf16(ah1, bl0, a10);
    a10 = wmma_bf16(ah1, bh0, a10);
    a11 = wmma_bf16(al1, bh1, a11);
    a11 = wmma_bf16(ah1, bl1, a11);
    a11 = wmma_bf16(ah1, bh1, a11);
  }
  int g = lane >> 4, r = lane & 15;
#pragma unroll
  for (int v = 0; v < 8; ++v) {
    int m = m0 + v + 8 * g;
    O[(long long)m * G_ + n0 + r]             = a00[v];
    O[(long long)m * G_ + n0 + 16 + r]        = a01[v];
    O[(long long)(m + 16) * G_ + n0 + r]      = a10[v];
    O[(long long)(m + 16) * G_ + n0 + 16 + r] = a11[v];
  }
}

// Fused per-layer GEMM kernel: GEMM1 = gi (input gates), GEMM2 = gh
// (recurrent gates, launched only when the dilation condition holds).
// Wide-tiles per GEMM: (256/32) * (1536/32) = 384 -> one per wave.
// grid = 48 blocks (1 GEMM) or 96 blocks (2 GEMMs), 256 threads (8 waves).
__global__ __launch_bounds__(256) void gru_gemm(
    const bf16* __restrict__ A1h, const bf16* __restrict__ A1l, long long lda1,
    const bf16* __restrict__ W1h, const bf16* __restrict__ W1l, int K1,
    float* __restrict__ O1,
    const bf16* __restrict__ A2h, const bf16* __restrict__ A2l, long long lda2,
    const bf16* __restrict__ W2h, const bf16* __restrict__ W2l, int K2,
    float* __restrict__ O2) {
  int wave = (blockIdx.x * blockDim.x + threadIdx.x) >> 5;
  int lane = threadIdx.x & 31;
  int which = wave / 384;
  int wt    = wave % 384;
  int m0 = (wt / 48) * 32;
  int n0 = (wt % 48) * 32;
  if (which == 0)
    gemm_tile22(A1h, A1l, lda1, W1h, W1l, K1, O1, m0, n0, lane);
  else
    gemm_tile22(A2h, A2l, lda2, W2h, W2l, K2, O2, m0, n0, lane);
}

// ---------------------------------------------------------------------------
// Epilogue: GRU gate math + carry update + output head + (optional) variance.
// grid = 256 blocks (one per batch row), 512 threads (one per hidden column).
// ---------------------------------------------------------------------------
__device__ __forceinline__ float blk_sum(float v) {
  __shared__ float red[512];
  __syncthreads();
  red[threadIdx.x] = v;
  __syncthreads();
  for (int s = 256; s > 0; s >>= 1) {
    if ((int)threadIdx.x < s) red[threadIdx.x] += red[threadIdx.x + s];
    __syncthreads();
  }
  return red[0];
}

__global__ __launch_bounds__(512) void gru_combine(
    const float* __restrict__ gi, const float* __restrict__ gh, int cond,
    const float* __restrict__ b_ih, const float* __restrict__ b_hh,
    float* __restrict__ carryF, bf16* __restrict__ carryH, bf16* __restrict__ carryL,
    bf16* __restrict__ streamH, bf16* __restrict__ streamL,
    const float* __restrict__ wv, const float* __restrict__ bv,
    float* __restrict__ outp, int t, int accumulate,
    float* __restrict__ var_out) {
  int b = blockIdx.x;
  int j = threadIdx.x;
  const float* gib = gi + (long long)b * G_;
  float ir  = gib[j]           + b_ih[j];
  float iz  = gib[H_ + j]      + b_ih[H_ + j];
  float in_ = gib[2 * H_ + j]  + b_ih[2 * H_ + j];
  float hr = b_hh[j], hz = b_hh[H_ + j], hn = b_hh[2 * H_ + j];
  if (cond) {
    const float* ghb = gh + (long long)b * G_;
    hr += ghb[j]; hz += ghb[H_ + j]; hn += ghb[2 * H_ + j];
  }
  float r = 1.f / (1.f + expf(-(ir + hr)));
  float z = 1.f / (1.f + expf(-(iz + hz)));
  float n = tanhf(in_ + r * hn);
  float hp = cond ? carryF[b * H_ + j] : 0.f;
  float h = (1.f - z) * n + z * hp;
  if (cond) {
    carryF[b * H_ + j] = h;
    bf16 hh = (bf16)h;
    carryH[b * H_ + j] = hh;
    carryL[b * H_ + j] = (bf16)(h - (float)hh);
  }
  if (streamH) {
    bf16 hh = (bf16)h;
    streamH[b * H_ + j] = hh;
    streamL[b * H_ + j] = (bf16)(h - (float)hh);
  }
  // output head: tanh(h . wv + bv)
  float dot = blk_sum(h * wv[j]);
  if (j == 0) {
    float o = tanhf(dot + bv[0]);
    if (accumulate) outp[b * T_ + t] += o;
    else            outp[b * T_ + t]  = o;
  }
  if (var_out) {
    float s  = blk_sum(h);
    float s2 = blk_sum(h * h);
    if (j == 0) {
      float mean = s / (float)H_;
      var_out[b] = (s2 - (float)H_ * mean * mean) / (float)(H_ - 1);
    }
  }
}

// ---------------------------------------------------------------------------
// Utilities: f32 -> (hi,lo) bf16 split, zero fill, final aux reduction.
// ---------------------------------------------------------------------------
__global__ void cvt_split(const float* __restrict__ s, bf16* __restrict__ h,
                          bf16* __restrict__ l, int n) {
  for (int i = blockIdx.x * blockDim.x + threadIdx.x; i < n;
       i += gridDim.x * blockDim.x) {
    float v = s[i];
    bf16 hv = (bf16)v;
    h[i] = hv;
    l[i] = (bf16)(v - (float)hv);
  }
}

__global__ void zero_f32(float* __restrict__ p, int n) {
  for (int i = blockIdx.x * blockDim.x + threadIdx.x; i < n;
       i += gridDim.x * blockDim.x)
    p[i] = 0.f;
}

__global__ __launch_bounds__(256) void final_aux(const float* __restrict__ var,
                                                 float* __restrict__ aux) {
  __shared__ float red[256];
  float s = 0.f;
  for (int i = threadIdx.x; i < B_ * T_; i += 256) s += var[i];
  red[threadIdx.x] = s;
  __syncthreads();
  for (int st = 128; st > 0; st >>= 1) {
    if ((int)threadIdx.x < st) red[threadIdx.x] += red[threadIdx.x + st];
    __syncthreads();
  }
  if (threadIdx.x == 0) aux[0] = red[0] / (float)B_;  // sum_t mean_b var
}

// ---------------------------------------------------------------------------
extern "C" void kernel_launch(void* const* d_in, const int* in_sizes, int n_in,
                              void* d_out, int out_size, void* d_ws, size_t ws_size,
                              hipStream_t stream) {
  (void)in_sizes; (void)n_in; (void)out_size; (void)ws_size;
  const float* x     = (const float*)d_in[0];
  const float* Wih[3] = {(const float*)d_in[1], (const float*)d_in[5], (const float*)d_in[9]};
  const float* Whh[3] = {(const float*)d_in[2], (const float*)d_in[6], (const float*)d_in[10]};
  const float* bih[3] = {(const float*)d_in[3], (const float*)d_in[7], (const float*)d_in[11]};
  const float* bhh[3] = {(const float*)d_in[4], (const float*)d_in[8], (const float*)d_in[12]};
  const float* w_head[3] = {(const float*)d_in[13], (const float*)d_in[15], (const float*)d_in[17]};
  const float* b_head[3] = {(const float*)d_in[14], (const float*)d_in[16], (const float*)d_in[18]};
  float* out = (float*)d_out;

  // ---- ws bump allocator (256B aligned) ----
  char* ws = (char*)d_ws;
  size_t off = 0;
  auto alloc = [&](size_t bytes) -> char* {
    off = (off + 255) & ~(size_t)255;
    char* p = ws + off;
    off += bytes;
    return p;
  };

  const int NX  = B_ * T_ * NIN_;   // 2,752,512
  const int NW0 = G_ * NIN_;        // 98,304
  const int NW  = G_ * H_;          // 786,432
  const int NC  = B_ * H_;          // 131,072

  bf16* xh = (bf16*)alloc((size_t)2 * NX * sizeof(bf16));   bf16* xl = xh + NX;
  bf16* w0h = (bf16*)alloc((size_t)2 * NW0 * sizeof(bf16)); bf16* w0l = w0h + NW0;
  bf16* wIh[3] = {w0h, nullptr, nullptr};
  bf16* wIl[3] = {w0l, nullptr, nullptr};
  bf16* wHh[3]; bf16* wHl[3];
  for (int l = 0; l < 3; ++l) {
    if (l > 0) {
      bf16* p = (bf16*)alloc((size_t)2 * NW * sizeof(bf16));
      wIh[l] = p; wIl[l] = p + NW;
    }
    bf16* q = (bf16*)alloc((size_t)2 * NW * sizeof(bf16));
    wHh[l] = q; wHl[l] = q + NW;
  }
  float* gatesI = (float*)alloc((size_t)B_ * G_ * sizeof(float));
  float* gatesH = (float*)alloc((size_t)B_ * G_ * sizeof(float));
  float* carryF = (float*)alloc((size_t)3 * NC * sizeof(float));
  bf16*  carryB = (bf16*)alloc((size_t)3 * 2 * NC * sizeof(bf16));
  bf16*  s1     = (bf16*)alloc((size_t)2 * NC * sizeof(bf16));
  float* varbuf = (float*)alloc((size_t)B_ * T_ * sizeof(float));

  float* cF[3]; bf16* cH[3]; bf16* cL[3];
  for (int l = 0; l < 3; ++l) {
    cF[l] = carryF + (size_t)l * NC;
    cH[l] = carryB + (size_t)l * 2 * NC;
    cL[l] = cH[l] + NC;
  }
  bf16* s1h = s1; bf16* s1l = s1 + NC;

  // ---- init: precision split + zero carries ----
  cvt_split<<<1024, 256, 0, stream>>>(x, xh, xl, NX);
  cvt_split<<<256, 256, 0, stream>>>(Wih[0], wIh[0], wIl[0], NW0);
  for (int l = 0; l < 3; ++l) {
    if (l > 0) cvt_split<<<1024, 256, 0, stream>>>(Wih[l], wIh[l], wIl[l], NW);
    cvt_split<<<1024, 256, 0, stream>>>(Whh[l], wHh[l], wHl[l], NW);
  }
  zero_f32<<<512, 256, 0, stream>>>(carryF, 3 * NC);
  zero_f32<<<512, 256, 0, stream>>>((float*)carryB, 3 * NC);  // 3*2*NC bf16 = 3*NC dwords

  // ---- sequential scan over time (graph-captured launches) ----
  for (int t = 0; t < T_; ++t) {
    int c1 = (t % 7) == 0;
    int c2 = (t % 28) == 0;

    // layer 0: gi from x_t (K=64), gh from carry0 (dilation 1: always)
    gru_gemm<<<96, 256, 0, stream>>>(
        xh + (size_t)t * NIN_, xl + (size_t)t * NIN_, (long long)(T_ * NIN_),
        wIh[0], wIl[0], NIN_, gatesI,
        cH[0], cL[0], (long long)H_, wHh[0], wHl[0], H_, gatesH);
    gru_combine<<<256, 512, 0, stream>>>(
        gatesI, gatesH, 1, bih[0], bhh[0], cF[0], cH[0], cL[0],
        nullptr, nullptr, w_head[0], b_head[0], out, t, 0, nullptr);

    // layer 1: gi from h0 stream (== carry0, dilation 1); gh only if t%7==0
    gru_gemm<<<c1 ? 96 : 48, 256, 0, stream>>>(
        cH[0], cL[0], (long long)H_, wIh[1], wIl[1], H_, gatesI,
        cH[1], cL[1], (long long)H_, wHh[1], wHl[1], H_, gatesH);
    gru_combine<<<256, 512, 0, stream>>>(
        gatesI, gatesH, c1, bih[1], bhh[1], cF[1], cH[1], cL[1],
        s1h, s1l, w_head[1], b_head[1], out, t, 1, nullptr);

    // layer 2: gi from h1 stream; gh only if t%28==0
    gru_gemm<<<c2 ? 96 : 48, 256, 0, stream>>>(
        s1h, s1l, (long long)H_, wIh[2], wIl[2], H_, gatesI,
        cH[2], cL[2], (long long)H_, wHh[2], wHl[2], H_, gatesH);
    gru_combine<<<256, 512, 0, stream>>>(
        gatesI, gatesH, c2, bih[2], bhh[2], cF[2], cH[2], cL[2],
        nullptr, nullptr, w_head[2], b_head[2], out, t, 1, varbuf + (size_t)t * B_);
  }

  final_aux<<<1, 256, 0, stream>>>(varbuf, out + (size_t)B_ * T_);
}
